// SelectiveBlock_61289183314184
// MI455X (gfx1250) — matrix operand order repaired and checked by reading, not compiled, not run on hardware
//
#include <hip/hip_runtime.h>
#include <hip/hip_bf16.h>

// ---------------------------------------------------------------------------
// Problem constants (match reference)
// ---------------------------------------------------------------------------
#define DIM_C   384
#define NHEAD   6
#define DHEAD   64
#define MLPD    1536
#define BATCH   8
#define SEQ     1024
#define TOK     (BATCH * SEQ)
#define QSCALE  0.125f          // DHEAD^-0.5

typedef __attribute__((ext_vector_type(16))) __bf16 v16bf;
typedef __attribute__((ext_vector_type(8)))  float  v8f;
typedef int vi4 __attribute__((vector_size(16)));   // int4 vector for async builtin

// ---------------------------------------------------------------------------
// Async global->LDS copies (CDNA5 GLOBAL_LOAD_ASYNC_TO_LDS_B128, ASYNCcnt),
// guarded so the kernel still compiles if the builtin is absent.
// Builtin signature (from hipcc diagnostic): param0 = AS1 (global) vi4*,
// param1 = LDS vi4*, then two immediates (offset, cpol).
// ---------------------------------------------------------------------------
#if defined(__has_builtin)
#if __has_builtin(__builtin_amdgcn_global_load_async_to_lds_b128)
#define ASYNC_LDS 1
#endif
#endif
#ifndef ASYNC_LDS
#define ASYNC_LDS 0
#endif

__device__ __forceinline__ void async_copy16(unsigned short* lds,
                                             const unsigned short* g) {
#if ASYNC_LDS
  // flat->AS pointer conversion via integer casts (LDS offset = low 32 bits,
  // global address = identity in flat space)
  __attribute__((address_space(1))) vi4* gp =
      (__attribute__((address_space(1))) vi4*)(unsigned long long)g;
  __attribute__((address_space(3))) vi4* lp =
      (__attribute__((address_space(3))) vi4*)(unsigned)(unsigned long long)lds;
  __builtin_amdgcn_global_load_async_to_lds_b128(gp, lp, 0, 0);
#else
  *(uint4*)lds = *(const uint4*)g;
#endif
}

__device__ __forceinline__ void async_wait_all() {
#if ASYNC_LDS
#if __has_builtin(__builtin_amdgcn_s_wait_asynccnt)
  __builtin_amdgcn_s_wait_asynccnt(0);
#else
  asm volatile("s_wait_asynccnt 0x0" ::: "memory");
#endif
#endif
}

// ---------------------------------------------------------------------------
// Helpers
// ---------------------------------------------------------------------------
__device__ __forceinline__ unsigned short f2bf(float f) {
  union { float f; unsigned u; } x; x.f = f;
  unsigned r = x.u + 0x7FFFu + ((x.u >> 16) & 1u);   // round-to-nearest-even
  return (unsigned short)(r >> 16);
}

// Load a 16-element bf16 fragment as two 16-byte chunks (LDS or global).
__device__ __forceinline__ v16bf load_frag16(const unsigned short* p0,
                                             const unsigned short* p1) {
  union { uint4 q[2]; v16bf v; } u;
  u.q[0] = *(const uint4*)p0;
  u.q[1] = *(const uint4*)p1;
  return u.v;
}

// ---------------------------------------------------------------------------
// f32 -> bf16 weight conversion (layout preserved: [out,in] row-major, which
// is exactly the WMMA B-operand friendly layout: B[k][n] = W[n][k])
// ---------------------------------------------------------------------------
__global__ void convert_bf16_kernel(const float* __restrict__ src,
                                    unsigned short* __restrict__ dst, int n) {
  int i = blockIdx.x * blockDim.x + threadIdx.x;
  if (i < n) dst[i] = f2bf(src[i]);
}

// ---------------------------------------------------------------------------
// LayerNorm over last dim (384), one wave (32 lanes) per row, bf16 output
// ---------------------------------------------------------------------------
__global__ __launch_bounds__(256) void ln_bf16_kernel(
    const float* __restrict__ x, const float* __restrict__ w,
    const float* __restrict__ b, unsigned short* __restrict__ out) {
  const int tid  = threadIdx.x;
  const int lane = tid & 31;
  const int wave = tid >> 5;
  const int row  = blockIdx.x * 8 + wave;
  const float* xr = x + (size_t)row * DIM_C;

  float v[12];
  float s = 0.f, sq = 0.f;
#pragma unroll
  for (int i = 0; i < 12; i++) {
    v[i] = xr[lane + 32 * i];
    s += v[i]; sq += v[i] * v[i];
  }
#pragma unroll
  for (int m = 1; m < 32; m <<= 1) {
    s  += __shfl_xor(s,  m, 32);
    sq += __shfl_xor(sq, m, 32);
  }
  const float mean = s * (1.0f / DIM_C);
  const float var  = sq * (1.0f / DIM_C) - mean * mean;
  const float rstd = rsqrtf(var + 1e-5f);

  unsigned short* orow = out + (size_t)row * DIM_C;
#pragma unroll
  for (int i = 0; i < 12; i++) {
    const int c = lane + 32 * i;
    orow[c] = f2bf((v[i] - mean) * rstd * w[c] + b[c]);
  }
}

// ---------------------------------------------------------------------------
// Tiled bf16 WMMA GEMM:  C[M,N] = A[M,K] @ Bw[N,K]^T   (Bw in [out,in] layout)
// Block tile 128x64, 8 waves, each wave 32x32 (2x2 WMMA 16x16x32 tiles).
// Double-buffered LDS pipeline: tile t+1 staged via async global->LDS copies
// while WMMAs consume tile t; one s_wait_asynccnt + barrier per k-step.
// EPI: 1 = +bias +residual -> f32 out
//      2 = +bias, exact GELU -> bf16 out
//      3 = QKV split epilogue -> q (scaled), k, v in [B,H,N,D] bf16
// ---------------------------------------------------------------------------
template <int EPI>
__global__ __launch_bounds__(256) void gemm_bf16_kernel(
    const unsigned short* __restrict__ A,
    const unsigned short* __restrict__ Bw,
    int M, int N, int K,
    const float* __restrict__ bias,
    const float* __restrict__ res,
    float* __restrict__ outF,
    unsigned short* __restrict__ outH,
    unsigned short* __restrict__ qo,
    unsigned short* __restrict__ ko,
    unsigned short* __restrict__ vo) {
  constexpr int LDA = 48;   // 96B row stride (16B aligned), dodges bank camping
  constexpr int LDB = 48;
  __shared__ __attribute__((aligned(16))) unsigned short sA[2][128 * LDA];
  __shared__ __attribute__((aligned(16))) unsigned short sB[2][64 * LDB];

  const int tid   = threadIdx.x;
  const int lane  = tid & 31;
  const int wave  = tid >> 5;
  const int wm    = wave & 3;     // 32-row quadrant
  const int wn    = wave >> 2;    // 32-col half
  const int bm0   = blockIdx.x * 128;
  const int bn0   = blockIdx.y * 64;
  const int l15   = lane & 15;
  const int lhalf = lane >> 4;

  v8f acc[2][2] = {};

  const int arow = tid >> 1, ahalf = tid & 1;   // A fill: 2 threads/row
  const int brow = tid >> 2, bq    = tid & 3;   // B fill: 4 threads/row

  const unsigned short* gA = A  + (size_t)(bm0 + arow) * K + ahalf * 16;
  const unsigned short* gB = Bw + (size_t)(bn0 + brow) * K + bq * 8;
  unsigned short* dA0 = &sA[0][arow * LDA + ahalf * 16];
  unsigned short* dA1 = &sA[1][arow * LDA + ahalf * 16];
  unsigned short* dB0 = &sB[0][brow * LDB + bq * 8];
  unsigned short* dB1 = &sB[1][brow * LDB + bq * 8];

  auto stage = [&](int k0, int buf) {
    unsigned short* da = buf ? dA1 : dA0;
    unsigned short* db = buf ? dB1 : dB0;
    async_copy16(da + 0, gA + k0 + 0);
    async_copy16(da + 8, gA + k0 + 8);
    async_copy16(db,     gB + k0);
  };

  stage(0, 0);                    // prologue
  async_wait_all();
  __syncthreads();

  const int nk = K >> 5;
  for (int t = 0; t < nk; t++) {
    const int cur = t & 1;
    if (t + 1 < nk) stage((t + 1) * 32, cur ^ 1);   // overlap with compute

    v16bf af[2], bfr[2];
#pragma unroll
    for (int i = 0; i < 2; i++) {
      // A layout: lane half selects K-subgroup; runs of 8 bf16 are contiguous
      const unsigned short* pr =
          &sA[cur][(wm * 32 + i * 16 + l15) * LDA + lhalf * 8];
      af[i] = load_frag16(pr, pr + 16);
    }
#pragma unroll
    for (int j = 0; j < 2; j++) {
      // B layout: lane half selects K-subgroup; runs of 8 bf16 are contiguous
      const unsigned short* pc =
          &sB[cur][(wn * 32 + j * 16 + l15) * LDB + lhalf * 8];
      bfr[j] = load_frag16(pc, pc + 16);
    }
#pragma unroll
    for (int i = 0; i < 2; i++)
#pragma unroll
      for (int j = 0; j < 2; j++)
        acc[i][j] = __builtin_amdgcn_wmma_f32_16x16x32_bf16(
            false, af[i], false, bfr[j], (short)0, acc[i][j], false, false);

    async_wait_all();   // next tile resident in LDS
    __syncthreads();    // all waves done with tile `cur`
  }

  // Epilogue. C layout: lane -> col = l15, rows r + 8*lhalf.
  const int gm_base = bm0 + wm * 32;
  const int gn_base = bn0 + wn * 32;
#pragma unroll
  for (int i = 0; i < 2; i++) {
#pragma unroll
    for (int j = 0; j < 2; j++) {
      const int gc = gn_base + j * 16 + l15;
#pragma unroll
      for (int r = 0; r < 8; r++) {
        const int gm = gm_base + i * 16 + r + 8 * lhalf;
        float v = acc[i][j][r];
        if (EPI == 1) {
          outF[(size_t)gm * N + gc] = v + bias[gc] + res[(size_t)gm * N + gc];
        } else if (EPI == 2) {
          float t = v + bias[gc];
          float g = 0.5f * t * (1.0f + erff(t * 0.70710678118654752f));
          outH[(size_t)gm * N + gc] = f2bf(g);
        } else {  // EPI == 3 : split into q/k/v [B,H,N,D] (uniform per block)
          int s  = gc / DIM_C;
          int rr = gc - s * DIM_C;
          int hh = rr >> 6;
          int dd = rr & 63;
          int bb = gm >> 10;
          int nn = gm & 1023;
          size_t dst = (((size_t)bb * NHEAD + hh) * SEQ + nn) * DHEAD + dd;
          if (s == 0)      qo[dst] = f2bf(v * QSCALE);
          else if (s == 1) ko[dst] = f2bf(v);
          else             vo[dst] = f2bf(v);
        }
      }
    }
  }
}

// ---------------------------------------------------------------------------
// Fused L2Q attention (flash-style, no softmax-max needed since the
// normalizer is a plain sum):
//   S = (Q*scale) K^T ; P = relu(a S^2 + b S + g) ; O = (sum_j P V) / (sum_j P)
// Block: 128 threads = 4 waves; wave owns 16 query rows of a 64-row block.
// K tile filled by async global->LDS; V transposed through VGPRs; next tile
// prefetched into cache with global_prefetch.
// ---------------------------------------------------------------------------
__global__ __launch_bounds__(128) void l2q_attn_kernel(
    const unsigned short* __restrict__ qh,   // [B,H,N,D] bf16, pre-scaled
    const unsigned short* __restrict__ kh,
    const unsigned short* __restrict__ vh,
    const float* __restrict__ alpha,
    const float* __restrict__ beta,
    const float* __restrict__ gamma,
    unsigned short* __restrict__ attno) {    // [B,N,H*D] bf16
  constexpr int LDK = 72, LDV = 72, LDP = 72;   // 144B strides, 16B aligned
  __shared__ __attribute__((aligned(16))) unsigned short sK [64 * LDK];
  __shared__ __attribute__((aligned(16))) unsigned short sVt[64 * LDV];
  __shared__ __attribute__((aligned(16))) unsigned short sP [4 * 16 * LDP];

  const int tid   = threadIdx.x;
  const int lane  = tid & 31;
  const int wave  = tid >> 5;
  const int l15   = lane & 15;
  const int lhalf = lane >> 4;
  const int bh    = blockIdx.y;             // b*NHEAD + h
  const int h     = bh % NHEAD;
  const int qi0   = blockIdx.x * 64;

  const float a_c = alpha[h], b_c = beta[h], g_c = gamma[h];
  const size_t base = (size_t)bh * SEQ * DHEAD;

  // Q fragments held in registers for the whole kernel (D=64 -> 2 K-steps)
  v16bf aq[2];
  {
    const unsigned short* qrow =
        qh + base + (size_t)(qi0 + wave * 16 + l15) * DHEAD;
#pragma unroll
    for (int s = 0; s < 2; s++) {
      const unsigned short* p = qrow + s * 32 + lhalf * 8;
      aq[s] = load_frag16(p, p + 16);
    }
  }

  v8f   oacc[4] = {};
  float dsum[8] = {0.f, 0.f, 0.f, 0.f, 0.f, 0.f, 0.f, 0.f};

  const int crow = tid >> 1, chalf = tid & 1;      // fill: 2 threads/row
  unsigned short* myP = &sP[wave * 16 * LDP];      // per-wave staging

  for (int j0 = 0; j0 < SEQ; j0 += 64) {
    __syncthreads();
    {
      // K tile: straight copy -> async global->LDS (ASYNCcnt tracked)
      const unsigned short* gk =
          kh + base + (size_t)(j0 + crow) * DHEAD + chalf * 32;
#pragma unroll
      for (int u = 0; u < 4; u++)
        async_copy16(&sK[crow * LDK + chalf * 32 + u * 8], gk + u * 8);

      // V tile: element transpose into [D][j] through VGPRs
      const unsigned short* gv =
          vh + base + (size_t)(j0 + crow) * DHEAD + chalf * 32;
#pragma unroll
      for (int u = 0; u < 4; u++) {
        union { uint4 q; unsigned short s[8]; } t;
        t.q = *(const uint4*)(gv + u * 8);
#pragma unroll
        for (int e = 0; e < 8; e++)
          sVt[(chalf * 32 + u * 8 + e) * LDV + crow] = t.s[e];
      }

      // warm L2/WGP$ for the next j-tile (global_prefetch_b8)
      if (j0 + 64 < SEQ) {
        __builtin_prefetch(gk + 64 * DHEAD, 0, 0);
        __builtin_prefetch(gv + 64 * DHEAD, 0, 0);
      }
    }
    async_wait_all();
    __syncthreads();

    // ---- S tiles (16x64 per wave), activation, stage P as A-operand ----
#pragma unroll
    for (int jt = 0; jt < 4; jt++) {
      v8f sacc = {};
#pragma unroll
      for (int s = 0; s < 2; s++) {
        const unsigned short* pk = &sK[(jt * 16 + l15) * LDK + s * 32 + lhalf * 8];
        v16bf bk = load_frag16(pk, pk + 16);
        sacc = __builtin_amdgcn_wmma_f32_16x16x32_bf16(
            false, aq[s], false, bk, (short)0, sacc, false, false);
      }
#pragma unroll
      for (int r = 0; r < 8; r++) {
        float xv = sacc[r];
        float pv = a_c * xv * xv + b_c * xv + g_c;
        pv = pv > 0.f ? pv : 0.f;
        dsum[r] += pv;
        myP[(r + 8 * lhalf) * LDP + jt * 16 + l15] = f2bf(pv);
      }
    }
    // per-wave LDS store->load ordering before consuming P as A fragments
    asm volatile("s_wait_dscnt 0x0" ::: "memory");

    // ---- O += P (16x64) @ V (64x64) ----
#pragma unroll
    for (int s2 = 0; s2 < 2; s2++) {
      const unsigned short* pp = &myP[l15 * LDP + s2 * 32 + lhalf * 8];
      v16bf ap = load_frag16(pp, pp + 16);
#pragma unroll
      for (int dt = 0; dt < 4; dt++) {
        const unsigned short* pv2 =
            &sVt[(dt * 16 + l15) * LDV + s2 * 32 + lhalf * 8];
        v16bf bv = load_frag16(pv2, pv2 + 16);
        oacc[dt] = __builtin_amdgcn_wmma_f32_16x16x32_bf16(
            false, ap, false, bv, (short)0, oacc[dt], false, false);
      }
    }
  }

  // Row sums live spread over each 16-lane half: xor-reduce, then normalize.
  const int b = bh / NHEAD;
#pragma unroll
  for (int r = 0; r < 8; r++) {
    float t = dsum[r];
    t += __shfl_xor(t, 1, 32);
    t += __shfl_xor(t, 2, 32);
    t += __shfl_xor(t, 4, 32);
    t += __shfl_xor(t, 8, 32);
    const float den = t + 1e-6f;
    const int row = qi0 + wave * 16 + r + 8 * lhalf;
    unsigned short* orow =
        attno + ((size_t)(b * SEQ + row)) * DIM_C + h * DHEAD;
#pragma unroll
    for (int dt = 0; dt < 4; dt++)
      orow[dt * 16 + l15] = f2bf(oacc[dt][r] / den);
  }
}

// ---------------------------------------------------------------------------
// Host-side orchestration
// ---------------------------------------------------------------------------
extern "C" void kernel_launch(void* const* d_in, const int* in_sizes, int n_in,
                              void* d_out, int out_size, void* d_ws,
                              size_t ws_size, hipStream_t stream) {
  (void)in_sizes; (void)n_in; (void)out_size; (void)ws_size;

  const float* x      = (const float*)d_in[0];
  const float* qkv_w  = (const float*)d_in[1];
  const float* proj_w = (const float*)d_in[2];
  const float* proj_b = (const float*)d_in[3];
  const float* alpha  = (const float*)d_in[4];
  const float* beta   = (const float*)d_in[5];
  const float* gamma  = (const float*)d_in[6];
  const float* ln1_w  = (const float*)d_in[7];
  const float* ln1_b  = (const float*)d_in[8];
  const float* ln2_w  = (const float*)d_in[9];
  const float* ln2_b  = (const float*)d_in[10];
  const float* w1     = (const float*)d_in[11];
  const float* b1     = (const float*)d_in[12];
  const float* w2     = (const float*)d_in[13];
  const float* b2     = (const float*)d_in[14];

  char* p = (char*)d_ws;
  auto carve = [&](size_t bytes) -> char* {
    char* r = p;
    p += (bytes + 255) & ~size_t(255);
    return r;
  };

  unsigned short* h1    = (unsigned short*)carve((size_t)TOK * DIM_C * 2);
  unsigned short* wqkv  = (unsigned short*)carve((size_t)3 * DIM_C * DIM_C * 2);
  unsigned short* wproj = (unsigned short*)carve((size_t)DIM_C * DIM_C * 2);
  unsigned short* w1b   = (unsigned short*)carve((size_t)MLPD * DIM_C * 2);
  unsigned short* w2b   = (unsigned short*)carve((size_t)DIM_C * MLPD * 2);
  unsigned short* qhb   = (unsigned short*)carve((size_t)TOK * DIM_C * 2);
  unsigned short* khb   = (unsigned short*)carve((size_t)TOK * DIM_C * 2);
  unsigned short* vhb   = (unsigned short*)carve((size_t)TOK * DIM_C * 2);
  unsigned short* attno = (unsigned short*)carve((size_t)TOK * DIM_C * 2);
  float*          x1    = (float*)carve((size_t)TOK * DIM_C * 4);
  unsigned short* h2    = (unsigned short*)carve((size_t)TOK * DIM_C * 2);
  unsigned short* mact  = (unsigned short*)carve((size_t)TOK * MLPD * 2);

  // 1) weights to bf16 (layout preserved)
  convert_bf16_kernel<<<(3 * DIM_C * DIM_C + 255) / 256, 256, 0, stream>>>(
      qkv_w, wqkv, 3 * DIM_C * DIM_C);
  convert_bf16_kernel<<<(DIM_C * DIM_C + 255) / 256, 256, 0, stream>>>(
      proj_w, wproj, DIM_C * DIM_C);
  convert_bf16_kernel<<<(MLPD * DIM_C + 255) / 256, 256, 0, stream>>>(
      w1, w1b, MLPD * DIM_C);
  convert_bf16_kernel<<<(DIM_C * MLPD + 255) / 256, 256, 0, stream>>>(
      w2, w2b, DIM_C * MLPD);

  // 2) LN1 -> bf16
  ln_bf16_kernel<<<TOK / 8, 256, 0, stream>>>(x, ln1_w, ln1_b, h1);

  // 3) QKV GEMM with split epilogue (Q pre-scaled by D^-0.5)
  gemm_bf16_kernel<3><<<dim3(TOK / 128, (3 * DIM_C) / 64), 256, 0, stream>>>(
      h1, wqkv, TOK, 3 * DIM_C, DIM_C, nullptr, nullptr, nullptr, nullptr,
      qhb, khb, vhb);

  // 4) fused L2Q attention
  l2q_attn_kernel<<<dim3(SEQ / 64, BATCH * NHEAD), 128, 0, stream>>>(
      qhb, khb, vhb, alpha, beta, gamma, attno);

  // 5) output projection + bias + residual -> x1 (f32)
  gemm_bf16_kernel<1><<<dim3(TOK / 128, DIM_C / 64), 256, 0, stream>>>(
      attno, wproj, TOK, DIM_C, DIM_C, proj_b, x, x1, nullptr,
      nullptr, nullptr, nullptr);

  // 6) LN2 -> bf16
  ln_bf16_kernel<<<TOK / 8, 256, 0, stream>>>(x1, ln2_w, ln2_b, h2);

  // 7) MLP up-proj + bias + exact GELU -> bf16
  gemm_bf16_kernel<2><<<dim3(TOK / 128, MLPD / 64), 256, 0, stream>>>(
      h2, w1b, TOK, MLPD, DIM_C, b1, nullptr, nullptr, mact,
      nullptr, nullptr, nullptr);

  // 8) MLP down-proj + bias + residual -> final output (f32)
  gemm_bf16_kernel<1><<<dim3(TOK / 128, DIM_C / 64), 256, 0, stream>>>(
      mact, w2b, TOK, DIM_C, MLPD, b2, x1, (float*)d_out,
      nullptr, nullptr, nullptr, nullptr);
}
